// FOFEReader_8005819039943
// MI455X (gfx1250) — compile-verified, matrix operand order checked by reading, not run on hardware
//
#include <hip/hip_runtime.h>
#include <hip/hip_bf16.h>
#include <math.h>

// ---------------------------------------------------------------------------
// Problem constants (match reference)
// ---------------------------------------------------------------------------
#define B_      32
#define LD_     512
#define LQ_     32
#define E_      300
#define NF_     4
#define D_      304      // E + NF
#define M_      16
#define ALPHA_  0.9f
#define H_      256
#define SAMPLE_ 256
#define POSN_   128
#define NEGN_   128
#define R_      16
#define PPL_    9        // NEG_N // R + 1
#define C3D_    912      // 3*D
#define CIN_    1212     // 3*D + E
#define KPAD1_  1216     // CIN padded to 32
#define NCOL_   (B_ * SAMPLE_)   // 8192

typedef __attribute__((ext_vector_type(16))) _Float16 v16h;
typedef __attribute__((ext_vector_type(8)))  _Float16 v8h;
typedef __attribute__((ext_vector_type(8)))  float    v8f;
typedef __attribute__((ext_vector_type(4)))  int      v4i;

// ---------------------------------------------------------------------------
// gfx1250 async global->LDS copy (ASYNCcnt path), with portable fallback.
// Builtin prototype (from toolchain diagnostic): 
//   void __builtin_amdgcn_global_load_async_to_lds_b128(
//        v4i addrspace(1)* src, v4i addrspace(3)* dst, imm offset, imm cpol)
// ---------------------------------------------------------------------------
#if __has_builtin(__builtin_amdgcn_global_load_async_to_lds_b128)
#define USE_ASYNC_LDS 1
#else
#define USE_ASYNC_LDS 0
#endif

__device__ __forceinline__ void stage16_to_lds(const _Float16* g, _Float16* l) {
#if USE_ASYNC_LDS
    __builtin_amdgcn_global_load_async_to_lds_b128(
        (__attribute__((address_space(1))) v4i*)(uintptr_t)g,
        (__attribute__((address_space(3))) v4i*)(uint32_t)(uintptr_t)l,
        0, 0);
#else
    *(v8h*)l = *(const v8h*)g;   // global_load_b128 + ds_store_b128
#endif
}

__device__ __forceinline__ void stage_wait() {
#if USE_ASYNC_LDS
#if __has_builtin(__builtin_amdgcn_s_wait_asynccnt)
    __builtin_amdgcn_s_wait_asynccnt(0);
#else
    asm volatile("s_wait_asynccnt 0x0" ::: "memory");
#endif
#endif
}

// ---------------------------------------------------------------------------
// FOFE helpers: x[b,c,t] = (c<E ? emb[doc[b,t],c] : doc_f[b,t,c-E])
// fwd(l,t) = sum_{k=0..l} alpha^k x[t-k] ; inv(l,t) = sum_{k=0..l} alpha^k x[t+k]
// ---------------------------------------------------------------------------
__device__ __forceinline__ float xval(const int* __restrict__ doc,
                                      const float* __restrict__ doc_f,
                                      const float* __restrict__ emb,
                                      int b, int c, int t) {
    if ((unsigned)t >= (unsigned)LD_) return 0.0f;
    if (c < E_) {
        int tok = doc[b * LD_ + t];
        return emb[(size_t)tok * E_ + c];
    }
    return doc_f[((size_t)b * LD_ + t) * NF_ + (c - E_)];
}

__device__ __forceinline__ float fofe_fwd(const int* doc, const float* doc_f, const float* emb,
                                          int b, int c, int l, int t) {
    float s = 0.0f, w = 1.0f;
    for (int k = 0; k <= l; ++k) { s += w * xval(doc, doc_f, emb, b, c, t - k); w *= ALPHA_; }
    return s;
}

__device__ __forceinline__ float fofe_inv(const int* doc, const float* doc_f, const float* emb,
                                          int b, int c, int l, int t) {
    float s = 0.0f, w = 1.0f;
    for (int k = 0; k <= l; ++k) { s += w * xval(doc, doc_f, emb, b, c, t + k); w *= ALPHA_; }
    return s;
}

// ---------------------------------------------------------------------------
// pos features: [B][912] = concat(l_ctx, ans, r_ctx) per channel block of 304
// ---------------------------------------------------------------------------
__global__ void pos_kernel(const int* __restrict__ doc, const float* __restrict__ doc_f,
                           const float* __restrict__ emb,
                           const int* __restrict__ target_s, const int* __restrict__ target_e,
                           float* __restrict__ pos) {
    int idx = blockIdx.x * blockDim.x + threadIdx.x;
    if (idx >= B_ * C3D_) return;
    int b = idx / C3D_;
    int rem = idx % C3D_;
    int part = rem / D_;
    int cc = rem % D_;
    int ts = target_s[b], te = target_e[b];
    int span = te - ts;
    float v;
    if (part == 0) {
        int t = ts - 1; if (t < 0) t = 0;
        v = fofe_fwd(doc, doc_f, emb, b, cc, M_ - 1, t);
    } else if (part == 1) {
        v = fofe_fwd(doc, doc_f, emb, b, cc, span, te);
    } else {
        int t = te + 1; if (t > LD_ - 1) t = LD_ - 1;
        v = fofe_inv(doc, doc_f, emb, b, cc, M_ - 1, t);
    }
    pos[idx] = v;
}

// ---------------------------------------------------------------------------
// negcat: [B][912][144]  (rand_cat in reference: concat(neg_l, neg_ans, neg_r))
// ---------------------------------------------------------------------------
__global__ void negcat_kernel(const int* __restrict__ doc, const float* __restrict__ doc_f,
                              const float* __restrict__ emb,
                              const int* __restrict__ rand_length,
                              const int* __restrict__ rand_position,
                              float* __restrict__ negcat) {
    int idx = blockIdx.x * blockDim.x + threadIdx.x;
    if (idx >= B_ * C3D_ * (R_ * PPL_)) return;
    int j = idx % (R_ * PPL_);
    int rem = idx / (R_ * PPL_);
    int b = rem / C3D_;
    int c = rem % C3D_;
    int part = c / D_;
    int cc = c % D_;
    int r = j / PPL_;
    int p = j % PPL_;
    int rl = rand_length[r];
    int rp = rand_position[r * PPL_ + p];
    float v;
    if (part == 0)      v = fofe_fwd(doc, doc_f, emb, b, cc, M_ - 1, rp - 1);
    else if (part == 1) v = fofe_fwd(doc, doc_f, emb, b, cc, rl, rp);
    else                v = fofe_inv(doc, doc_f, emb, b, cc, M_ - 1, rp + rl + 1);
    negcat[idx] = v;
}

// ---------------------------------------------------------------------------
// qf[b,e] = sum_t alpha^(LQ-1-t) * emb[query[b,t], e]
// ---------------------------------------------------------------------------
__global__ void qf_kernel(const int* __restrict__ query, const float* __restrict__ emb,
                          float* __restrict__ qf) {
    int idx = blockIdx.x * blockDim.x + threadIdx.x;
    if (idx >= B_ * E_) return;
    int b = idx / E_;
    int e = idx % E_;
    float s = 0.0f, w = 1.0f;
    for (int t = LQ_ - 1; t >= 0; --t) {
        s += w * emb[(size_t)query[b * LQ_ + t] * E_ + e];
        w *= ALPHA_;
    }
    qf[idx] = s;
}

// ---------------------------------------------------------------------------
// Assemble X0 as f16, layout [N=8192][KPAD1=1216], K contiguous (WMMA B layout)
// ---------------------------------------------------------------------------
__global__ void assemble_kernel(const float* __restrict__ pos, const float* __restrict__ negcat,
                                const float* __restrict__ qf,
                                const int* __restrict__ rand_idx, const int* __restrict__ perm_idx,
                                _Float16* __restrict__ X0) {
    int idx = blockIdx.x * blockDim.x + threadIdx.x;
    if (idx >= NCOL_ * KPAD1_) return;
    int n = idx / KPAD1_;
    int k = idx % KPAD1_;
    int b = n >> 8;
    int s = n & 255;
    float v = 0.0f;
    if (k < C3D_) {
        int ps = perm_idx[s];
        if (ps < POSN_) {
            v = pos[b * C3D_ + k];
        } else {
            int j = rand_idx[ps - POSN_];          // index into 144 rand columns
            v = negcat[((size_t)(b * C3D_ + k)) * (R_ * PPL_) + j];
        }
    } else if (k < CIN_) {
        v = qf[b * E_ + (k - C3D_)];
    } // else zero pad
    X0[idx] = (_Float16)v;
}

// ---------------------------------------------------------------------------
// Weight conversion: f32 [O][I] -> f16 [Opad][Ipad], zero padded
// ---------------------------------------------------------------------------
__global__ void wconv_kernel(const float* __restrict__ W, _Float16* __restrict__ Wh,
                             int O, int I, int Opad, int Ipad) {
    int idx = blockIdx.x * blockDim.x + threadIdx.x;
    if (idx >= Opad * Ipad) return;
    int o = idx / Ipad;
    int i = idx % Ipad;
    float v = (o < O && i < I) ? W[(size_t)o * I + i] : 0.0f;
    Wh[idx] = (_Float16)v;
}

// ---------------------------------------------------------------------------
// WMMA GEMM with double-buffered async LDS staging of the activation tile.
//   A  : f16 [Mtotal][K]   (weights, K contiguous)     -> direct global loads
//   Bm : f16 [N][K]        (activations, K contiguous) -> async staged in LDS
//   Y  : f16 [N][Mstride]  (written transposed: next layer's Bm)
// Block: 8 waves = 4(M) x 2(N); wave tile 64x32; macro 256(M) x 64(N).
// Per K-step(32): block stages one 64x32 B tile (4KB, 1 b128/thread, ASYNCcnt),
// each wave does 8x v_wmma_f32_16x16x32_f16 reading B frags from LDS.
// ---------------------------------------------------------------------------
#define BSTRIDE 40   // halves per LDS row (80B pitch: 16B aligned, bank-spread)

__global__ __launch_bounds__(256) void gemm_wmma_kernel(
        const _Float16* __restrict__ A, const _Float16* __restrict__ Bm,
        _Float16* __restrict__ Y, int Mtotal, int K, int Mstride, int relu) {
    __shared__ _Float16 lbuf[2][64 * BSTRIDE];     // 10 KB double buffer

    const int tid  = threadIdx.x;
    const int wave = tid >> 5;
    const int lane = tid & 31;
    const int mi = wave & 3;                 // 4 wave rows (M)
    const int nj = wave >> 2;                // 2 wave cols (N)
    const int mbase = blockIdx.x * 256 + mi * 64;
    const int nblk  = blockIdx.y * 64;       // block's N base
    const int nloc  = nj * 32;               // wave's N base within block
    const int rsel  = lane & 15;             // row (A) / col (B) within 16-tile
    const int khalf = (lane >> 4) << 3;      // lanes 16-31 carry K 8-15 / 24-31
    const bool active = (mbase < Mtotal);    // wave-uniform (EXEC stays full)

    // stage-loader mapping: thread -> one 16B chunk of the 64(col) x 32(k) tile
    const int lcol = tid >> 2;               // 0..63
    const int lchk = (tid & 3) << 3;         // k offset in halves: 0,8,16,24
    const _Float16* gsrc = Bm + (size_t)(nblk + lcol) * K + lchk;
    _Float16* ldst[2] = { &lbuf[0][lcol * BSTRIDE + lchk],
                          &lbuf[1][lcol * BSTRIDE + lchk] };

    v8f acc[4][2];
    const v8f vzero = {};
    #pragma unroll
    for (int t = 0; t < 4; ++t)
        #pragma unroll
        for (int u = 0; u < 2; ++u) acc[t][u] = vzero;

    // preload first tile
    stage16_to_lds(gsrc, ldst[0]);
    stage_wait();
    __syncthreads();

    int buf = 0;
    for (int k0 = 0; k0 < K; k0 += 32) {
        const bool more = (k0 + 32 < K);
        if (more)   // async DMA of next tile overlaps this step's WMMAs
            stage16_to_lds(gsrc + k0 + 32, ldst[buf ^ 1]);

        if (active) {
            v16h afrag[4], bfrag[2];
            #pragma unroll
            for (int t = 0; t < 4; ++t) {
                const _Float16* ap =
                    A + (size_t)(mbase + t * 16 + rsel) * K + k0 + khalf;
                v8h lo = *(const v8h*)ap;          // K chunk +0..7 / +8..15
                v8h hi = *(const v8h*)(ap + 16);   // K chunk +16..23 / +24..31
                afrag[t] = __builtin_shufflevector(lo, hi, 0,1,2,3,4,5,6,7,
                                                           8,9,10,11,12,13,14,15);
            }
            #pragma unroll
            for (int u = 0; u < 2; ++u) {
                const _Float16* bp =
                    &lbuf[buf][(nloc + u * 16 + rsel) * BSTRIDE + khalf];
                v8h lo = *(const v8h*)bp;          // ds_load_b128
                v8h hi = *(const v8h*)(bp + 16);
                bfrag[u] = __builtin_shufflevector(lo, hi, 0,1,2,3,4,5,6,7,
                                                           8,9,10,11,12,13,14,15);
            }
            if (more)  // global_prefetch_b8 for next weight chunk
                __builtin_prefetch((const void*)(A + (size_t)(mbase + rsel) * K + k0 + 32), 0, 3);

            #pragma unroll
            for (int t = 0; t < 4; ++t)
                #pragma unroll
                for (int u = 0; u < 2; ++u)
                    acc[t][u] = __builtin_amdgcn_wmma_f32_16x16x32_f16(
                        false, afrag[t], false, bfrag[u],
                        (short)0, acc[t][u], false, false);
        }

        if (more) {
            stage_wait();        // own async load landed in LDS
            __syncthreads();     // whole tile visible; old buffer free to overwrite
            buf ^= 1;
        }
    }

    if (!active) return;

    // Epilogue: lane holds rows (tile_m + khalf + 0..7) of column (tile_n + rsel)
    #pragma unroll
    for (int t = 0; t < 4; ++t) {
        #pragma unroll
        for (int u = 0; u < 2; ++u) {
            const int n      = nblk + nloc + u * 16 + rsel;
            const int mstart = mbase + t * 16 + khalf;
            v8h out;
            #pragma unroll
            for (int j = 0; j < 8; ++j) {
                float v = acc[t][u][j];
                if (relu) v = v > 0.0f ? v : 0.0f;
                out[j] = (_Float16)v;
            }
            *(v8h*)(Y + (size_t)n * Mstride + mstart) = out;
        }
    }
}

// ---------------------------------------------------------------------------
// Loss: sum over 8192 columns of (sigmoid(logit) - target)^2
// ---------------------------------------------------------------------------
__global__ void zero_kernel(float* out) { if (threadIdx.x == 0 && blockIdx.x == 0) out[0] = 0.0f; }

__global__ void loss_kernel(const _Float16* __restrict__ logits, int Mstride,
                            const int* __restrict__ perm_idx, float* __restrict__ out) {
    int idx = blockIdx.x * blockDim.x + threadIdx.x;
    if (idx >= NCOL_) return;
    int s = idx & 255;
    float z = (float)logits[(size_t)idx * Mstride + 0];
    float sc = 1.0f / (1.0f + expf(-z));
    float tgt = (perm_idx[s] < POSN_) ? 1.0f : 0.0f;
    float d = sc - tgt;
    atomicAdd(out, d * d);
}

// ---------------------------------------------------------------------------
// Host launch
// ---------------------------------------------------------------------------
extern "C" void kernel_launch(void* const* d_in, const int* in_sizes, int n_in,
                              void* d_out, int out_size, void* d_ws, size_t ws_size,
                              hipStream_t stream) {
    (void)in_sizes; (void)n_in; (void)out_size; (void)ws_size;
    const int*   doc      = (const int*)  d_in[0];
    const float* doc_f    = (const float*)d_in[1];
    const int*   query    = (const int*)  d_in[2];
    const int*   target_s = (const int*)  d_in[3];
    const int*   target_e = (const int*)  d_in[4];
    // d_in[5], d_in[6] = masks (unused by reference)
    const float* emb      = (const float*)d_in[7];
    const float* W1       = (const float*)d_in[8];
    const float* W2       = (const float*)d_in[9];
    const float* W3       = (const float*)d_in[10];
    const float* W4       = (const float*)d_in[11];
    const float* W5       = (const float*)d_in[12];
    const int*   rand_len = (const int*)  d_in[13];
    const int*   rand_pos = (const int*)  d_in[14];
    const int*   rand_idx = (const int*)  d_in[15];
    const int*   perm_idx = (const int*)  d_in[16];

    // workspace carve-up (256B aligned)
    char* base = (char*)d_ws;
    size_t off = 0;
    auto carve = [&](size_t bytes) -> void* {
        void* p = base + off;
        off = (off + bytes + 255) & ~(size_t)255;
        return p;
    };
    float*    pos    = (float*)   carve((size_t)B_ * C3D_ * 4);                 // 117 KB
    float*    qf     = (float*)   carve((size_t)B_ * E_ * 4);                   // 38 KB
    float*    negcat = (float*)   carve((size_t)B_ * C3D_ * R_ * PPL_ * 4);     // 16.8 MB
    _Float16* XA     = (_Float16*)carve((size_t)NCOL_ * KPAD1_ * 2);            // 19.9 MB
    _Float16* XB     = (_Float16*)carve((size_t)NCOL_ * 1024 * 2);              // 16.8 MB
    _Float16* W1h    = (_Float16*)carve((size_t)1024 * KPAD1_ * 2);             // 2.5 MB
    _Float16* W2h    = (_Float16*)carve((size_t)1024 * 1024 * 2);               // 2.1 MB
    _Float16* W3h    = (_Float16*)carve((size_t)1024 * 1024 * 2);               // 2.1 MB
    _Float16* W4h    = (_Float16*)carve((size_t)512  * 1024 * 2);               // 1.0 MB
    _Float16* W5h    = (_Float16*)carve((size_t)64   * 512  * 2);               // 64 KB
    _Float16* logits = (_Float16*)carve((size_t)NCOL_ * 64 * 2);                // 1.0 MB

    const int TB = 256;
    auto blocks = [&](long long n) { return dim3((unsigned)((n + TB - 1) / TB)); };

    // --- feature extraction ---
    pos_kernel<<<blocks((long long)B_ * C3D_), TB, 0, stream>>>(doc, doc_f, emb, target_s, target_e, pos);
    qf_kernel<<<blocks((long long)B_ * E_), TB, 0, stream>>>(query, emb, qf);
    negcat_kernel<<<blocks((long long)B_ * C3D_ * R_ * PPL_), TB, 0, stream>>>(
        doc, doc_f, emb, rand_len, rand_pos, negcat);
    assemble_kernel<<<blocks((long long)NCOL_ * KPAD1_), TB, 0, stream>>>(
        pos, negcat, qf, rand_idx, perm_idx, XA);

    // --- weight conversion to f16 (zero padded) ---
    wconv_kernel<<<blocks((long long)1024 * KPAD1_), TB, 0, stream>>>(W1, W1h, 1024, CIN_, 1024, KPAD1_);
    wconv_kernel<<<blocks((long long)1024 * 1024), TB, 0, stream>>>(W2, W2h, 1024, 1024, 1024, 1024);
    wconv_kernel<<<blocks((long long)1024 * 1024), TB, 0, stream>>>(W3, W3h, 1024, 1024, 1024, 1024);
    wconv_kernel<<<blocks((long long)512 * 1024), TB, 0, stream>>>(W4, W4h, 512, 1024, 512, 1024);
    wconv_kernel<<<blocks((long long)64 * 512), TB, 0, stream>>>(W5, W5h, 1, 512, 64, 512);

    // --- MLP via WMMA GEMMs (output written [N][M] = next layer's input layout) ---
    dim3 gblk(256);
    // L1: [1024 x 1216] x [1216 x 8192] -> XB [N][1024], relu
    gemm_wmma_kernel<<<dim3(1024 / 256, NCOL_ / 64), gblk, 0, stream>>>(W1h, XA, XB, 1024, KPAD1_, 1024, 1);
    // L2: [1024 x 1024] x XB -> XA (reused) [N][1024], relu
    gemm_wmma_kernel<<<dim3(1024 / 256, NCOL_ / 64), gblk, 0, stream>>>(W2h, XB, XA, 1024, 1024, 1024, 1);
    // L3: [1024 x 1024] x XA -> XB [N][1024], relu
    gemm_wmma_kernel<<<dim3(1024 / 256, NCOL_ / 64), gblk, 0, stream>>>(W3h, XA, XB, 1024, 1024, 1024, 1);
    // L4: [512 x 1024] x XB -> XA [N][512], relu
    gemm_wmma_kernel<<<dim3(512 / 256, NCOL_ / 64), gblk, 0, stream>>>(W4h, XB, XA, 512, 1024, 512, 1);
    // L5: [64(pad of 1) x 512] x XA -> logits [N][64], linear
    gemm_wmma_kernel<<<dim3(1, NCOL_ / 64), gblk, 0, stream>>>(W5h, XA, logits, 64, 512, 64, 0);

    // --- loss ---
    zero_kernel<<<1, 32, 0, stream>>>((float*)d_out);
    loss_kernel<<<blocks(NCOL_), TB, 0, stream>>>(logits, 64, perm_idx, (float*)d_out);
}
